// Attention3d_56375740727850
// MI455X (gfx1250) — matrix operand order repaired and checked by reading, not compiled, not run on hardware
//
#include <hip/hip_runtime.h>
#include <hip/hip_bf16.h>

typedef __attribute__((ext_vector_type(16))) __bf16 v16bf;
typedef __attribute__((ext_vector_type(8)))  float  v8f;

#define WMMA_BF16(A, B, C) \
  __builtin_amdgcn_wmma_f32_16x16x32_bf16(false, (A), false, (B), (short)0, (C), false, false)

// Problem constants
#define BB   8
#define CC   128
#define CA   16
#define CV   64
#define NQ   13824   // 24*24*24
#define NK   1728    // 12*12*12
#define PROJ_ROWS 96 // 16 q + 16 k + 64 v

// ---------------------------------------------------------------------------
// Kernel 0: convert weights to bf16. wbf = [wq;wk;wv] row-major [96][128],
// wobf = w_o row-major [128][64].
// ---------------------------------------------------------------------------
__global__ void attn3d_wconv(const float* __restrict__ wq, const float* __restrict__ wk,
                             const float* __restrict__ wv, const float* __restrict__ wo,
                             __bf16* __restrict__ wbf, __bf16* __restrict__ wobf) {
  const int total = PROJ_ROWS * CC + CC * CV;
  for (int i = blockIdx.x * blockDim.x + threadIdx.x; i < total; i += gridDim.x * blockDim.x) {
    if (i < PROJ_ROWS * CC) {
      int row = i >> 7, c = i & 127;
      float v = (row < 16) ? wq[(row << 7) + c]
              : (row < 32) ? wk[((row - 16) << 7) + c]
                           : wv[((row - 32) << 7) + c];
      wbf[i] = (__bf16)v;
    } else {
      int j = i - PROJ_ROWS * CC;
      wobf[j] = (__bf16)wo[j];
    }
  }
}

// ---------------------------------------------------------------------------
// Kernel 1: projection GEMM. projall[b][row][n] = sum_c wbf[row][c]*x[b][c][n]
// 4 waves/block, 32 columns per wave (128 per block). 6 M-tiles x 4 K-chunks.
// ---------------------------------------------------------------------------
__global__ __launch_bounds__(128) void attn3d_proj(const float* __restrict__ x,
                                                   const __bf16* __restrict__ wbf,
                                                   __bf16* __restrict__ projall) {
  __shared__ __align__(16) __bf16 sW[PROJ_ROWS * CC];  // 24 KB

  const int tid  = threadIdx.x;
  const int wid  = tid >> 5;
  const int lane = tid & 31;
  const int lo = lane & 15, hi = lane >> 4;
  const int b  = blockIdx.y;
  const int n0 = blockIdx.x * 128 + wid * 32;

  // stage weights: 24576 B = 1536 uint4, 12 per thread
  {
    const uint4* src = (const uint4*)wbf;
    uint4* dst = (uint4*)sW;
#pragma unroll
    for (int i = 0; i < 12; ++i) dst[tid + 128 * i] = src[tid + 128 * i];
  }
  __syncthreads();

  v8f zero = {};
  v8f acc[6][2];
#pragma unroll
  for (int m = 0; m < 6; ++m) { acc[m][0] = zero; acc[m][1] = zero; }

#pragma unroll
  for (int cc = 0; cc < 4; ++cc) {
    // B fragments: x tiles [32 c x 16 n] x2, fp32 -> bf16
    v16bf bx0, bx1;
#pragma unroll
    for (int t = 0; t < 16; ++t) {
      int c = cc * 32 + t + 16 * hi;                       // B layout K index
      const float* src = x + (size_t)(b * CC + c) * NQ + n0 + lo;
      bx0[t] = (__bf16)src[0];
      bx1[t] = (__bf16)src[16];
    }
#pragma unroll
    for (int mt = 0; mt < 6; ++mt) {
      v16bf aw;
#pragma unroll
      for (int t = 0; t < 16; ++t) {
        int kk = t + 16 * hi;                                 // A layout K index
        aw[t] = sW[(mt * 16 + lo) * CC + cc * 32 + kk];
      }
      acc[mt][0] = WMMA_BF16(aw, bx0, acc[mt][0]);
      acc[mt][1] = WMMA_BF16(aw, bx1, acc[mt][1]);
    }
  }

#pragma unroll
  for (int mt = 0; mt < 6; ++mt)
#pragma unroll
    for (int sub = 0; sub < 2; ++sub)
#pragma unroll
      for (int r = 0; r < 8; ++r) {
        int row = mt * 16 + r + 8 * hi;
        projall[(size_t)(b * PROJ_ROWS + row) * NQ + n0 + sub * 16 + lo] =
            (__bf16)acc[mt][sub][r];
      }
}

// ---------------------------------------------------------------------------
// Kernel 2: 2x2x2 maxpool of k/v rows of projall -> kbuf [8,16,1728],
// vbuf [8,64,1728] (bf16; rounding commutes with max on bf16 inputs).
// ---------------------------------------------------------------------------
__global__ void attn3d_pool(const __bf16* __restrict__ projall,
                            __bf16* __restrict__ kbuf, __bf16* __restrict__ vbuf) {
  int idx = blockIdx.x * blockDim.x + threadIdx.x;
  const int total = BB * 80 * NK;
  if (idx >= total) return;
  int p  = idx % NK;
  int t  = idx / NK;
  int ch = t % 80;
  int b  = t / 80;
  int w2 = p % 12, h2 = (p / 12) % 12, l2 = p / 144;
  const __bf16* src = projall + (size_t)(b * PROJ_ROWS + 16 + ch) * NQ;
  float m = -3.0e38f;
#pragma unroll
  for (int dl = 0; dl < 2; ++dl)
#pragma unroll
    for (int dh = 0; dh < 2; ++dh)
#pragma unroll
      for (int dw = 0; dw < 2; ++dw) {
        int n = (2 * l2 + dl) * 576 + (2 * h2 + dh) * 24 + (2 * w2 + dw);
        m = fmaxf(m, (float)src[n]);
      }
  __bf16 mv = (__bf16)m;
  if (ch < 16) kbuf[(size_t)(b * CA + ch) * NK + p] = mv;
  else         vbuf[(size_t)(b * CV + (ch - 16)) * NK + p] = mv;
}

// ---------------------------------------------------------------------------
// Kernel 3: flash attention + fused output projection + residual.
// 4 waves/block sharing K/V staging; each wave owns a 32-query stripe.
// ---------------------------------------------------------------------------
struct LoopBufs {
  __bf16 k[CA * 32];       // 1 KB   [c][kk]
  __bf16 v[CV * 32];       // 4 KB   [cv][kk]
  float  p[4][32 * 32];    // 16 KB  per-wave [kk][n]  (n: 0-15 tile0, 16-31 tile1)
};
union ShBufs {
  LoopBufs loop;
  float o[4][CV * 32];     // 32 KB  per-wave [cv][n]
};

__global__ __launch_bounds__(128) void attn3d_flash(const __bf16* __restrict__ projall,
                                                    const __bf16* __restrict__ kbuf,
                                                    const __bf16* __restrict__ vbuf,
                                                    const __bf16* __restrict__ wobf,
                                                    const float* __restrict__ x,
                                                    const float* __restrict__ gamma,
                                                    float* __restrict__ out) {
  __shared__ __align__(16) __bf16 sWo[CC * CV];  // 16 KB persistent
  __shared__ __align__(16) ShBufs u;             // 32 KB (loop / epilogue overlap)

  const int tid  = threadIdx.x;
  const int wid  = tid >> 5;
  const int lane = tid & 31;
  const int lo = lane & 15, hi = lane >> 4;
  const int b  = blockIdx.y;
  const int n0 = blockIdx.x * 128 + wid * 32;

  // stage Wo (16384 B = 1024 uint4, 8 per thread)
  {
    const uint4* src = (const uint4*)wobf;
    uint4* dst = (uint4*)sWo;
#pragma unroll
    for (int i = 0; i < 8; ++i) dst[tid + 128 * i] = src[tid + 128 * i];
  }

  // persistent Q B-fragments: [32 c x 16 n] x2, channels 16..31 zero-padded
  v16bf bq0, bq1;
#pragma unroll
  for (int t = 0; t < 16; ++t) {
    int c = t + 16 * hi;
    if (c < CA) {
      const __bf16* src = projall + (size_t)(b * PROJ_ROWS + c) * NQ + n0 + lo;
      bq0[t] = src[0];
      bq1[t] = src[16];
    } else {
      bq0[t] = (__bf16)0.0f;
      bq1[t] = (__bf16)0.0f;
    }
  }

  v8f zero = {};
  v8f acc00 = zero, acc01 = zero, acc02 = zero, acc03 = zero;  // n-tile 0, cv tiles
  v8f acc10 = zero, acc11 = zero, acc12 = zero, acc13 = zero;  // n-tile 1
  float mrun0 = -3.0e38f, lrun0 = 0.0f;
  float mrun1 = -3.0e38f, lrun1 = 0.0f;

  float* sPw = u.loop.p[wid];

  __syncthreads();

  for (int kbase = 0; kbase < NK; kbase += 32) {
    // stage K chunk [16 x 32] (64 uint4) and V chunk [64 x 32] (256 uint4)
    if (tid < 64) {
      int row = tid >> 2, seg = tid & 3;
      *(uint4*)(u.loop.k + row * 32 + seg * 8) =
          *(const uint4*)(kbuf + (size_t)(b * CA + row) * NK + kbase + seg * 8);
    }
#pragma unroll
    for (int i = 0; i < 2; ++i) {
      int idx = tid + 128 * i;
      int row = idx >> 2, seg = idx & 3;
      *(uint4*)(u.loop.v + row * 32 + seg * 8) =
          *(const uint4*)(vbuf + (size_t)(b * CV + row) * NK + kbase + seg * 8);
    }
    __syncthreads();

    // A fragments of K^T: [M=key row, K=channel] (channels >=16 zero)
    v16bf ak0, ak1;
#pragma unroll
    for (int t = 0; t < 16; ++t) {
      int c = t + 16 * hi;
      ak0[t] = (c < CA) ? u.loop.k[c * 32 + lo]      : (__bf16)0.0f;
      ak1[t] = (c < CA) ? u.loop.k[c * 32 + 16 + lo] : (__bf16)0.0f;
    }
    v8f s0 = WMMA_BF16(ak0, bq0, zero);   // keys +0..15,  queries n0+0..15
    v8f s1 = WMMA_BF16(ak1, bq0, zero);   // keys +16..31, queries n0+0..15
    v8f s2 = WMMA_BF16(ak0, bq1, zero);   // keys +0..15,  queries n0+16..31
    v8f s3 = WMMA_BF16(ak1, bq1, zero);   // keys +16..31, queries n0+16..31

    // online softmax, n-tile 0 (column lives in lane pair {lane, lane^16})
    {
      float cmax = -3.0e38f;
#pragma unroll
      for (int r = 0; r < 8; ++r) { cmax = fmaxf(cmax, s0[r]); cmax = fmaxf(cmax, s1[r]); }
      cmax = fmaxf(cmax, __shfl_xor(cmax, 16, 32));
      float mnew = fmaxf(mrun0, cmax);
      float corr = __expf(mrun0 - mnew);
      float psum = 0.0f;
#pragma unroll
      for (int r = 0; r < 8; ++r) {
        float p0 = __expf(s0[r] - mnew);
        float p1 = __expf(s1[r] - mnew);
        psum += p0 + p1;
        sPw[(r + 8 * hi) * 32 + lo]      = p0;
        sPw[(16 + r + 8 * hi) * 32 + lo] = p1;
      }
      psum += __shfl_xor(psum, 16, 32);
      lrun0 = lrun0 * corr + psum;
      mrun0 = mnew;
      acc00 = acc00 * corr; acc01 = acc01 * corr; acc02 = acc02 * corr; acc03 = acc03 * corr;
    }
    // online softmax, n-tile 1
    {
      float cmax = -3.0e38f;
#pragma unroll
      for (int r = 0; r < 8; ++r) { cmax = fmaxf(cmax, s2[r]); cmax = fmaxf(cmax, s3[r]); }
      cmax = fmaxf(cmax, __shfl_xor(cmax, 16, 32));
      float mnew = fmaxf(mrun1, cmax);
      float corr = __expf(mrun1 - mnew);
      float psum = 0.0f;
#pragma unroll
      for (int r = 0; r < 8; ++r) {
        float p0 = __expf(s2[r] - mnew);
        float p1 = __expf(s3[r] - mnew);
        psum += p0 + p1;
        sPw[(r + 8 * hi) * 32 + 16 + lo]      = p0;
        sPw[(16 + r + 8 * hi) * 32 + 16 + lo] = p1;
      }
      psum += __shfl_xor(psum, 16, 32);
      lrun1 = lrun1 * corr + psum;
      mrun1 = mnew;
      acc10 = acc10 * corr; acc11 = acc11 * corr; acc12 = acc12 * corr; acc13 = acc13 * corr;
    }

    // restripe P into B-fragments [32 k x 16 n] x2 (within-wave LDS ordering)
    v16bf bp0, bp1;
#pragma unroll
    for (int t = 0; t < 16; ++t) {
      int kk = t + 16 * hi;
      bp0[t] = (__bf16)sPw[kk * 32 + lo];
      bp1[t] = (__bf16)sPw[kk * 32 + 16 + lo];
    }

    // O += V * P, four 16-row cv tiles x two n-tiles
#pragma unroll
    for (int cvt = 0; cvt < 4; ++cvt) {
      v16bf av;
#pragma unroll
      for (int t = 0; t < 16; ++t) {
        int kk = t + 16 * hi;
        av[t] = u.loop.v[(cvt * 16 + lo) * 32 + kk];
      }
      if      (cvt == 0) { acc00 = WMMA_BF16(av, bp0, acc00); acc10 = WMMA_BF16(av, bp1, acc10); }
      else if (cvt == 1) { acc01 = WMMA_BF16(av, bp0, acc01); acc11 = WMMA_BF16(av, bp1, acc11); }
      else if (cvt == 2) { acc02 = WMMA_BF16(av, bp0, acc02); acc12 = WMMA_BF16(av, bp1, acc12); }
      else               { acc03 = WMMA_BF16(av, bp0, acc03); acc13 = WMMA_BF16(av, bp1, acc13); }
    }
    __syncthreads();
  }

  // normalize and restripe O through per-wave LDS region (aliases loop bufs)
  float* sOw = u.o[wid];
  float inv0 = 1.0f / lrun0;
  float inv1 = 1.0f / lrun1;
#pragma unroll
  for (int r = 0; r < 8; ++r) {
    sOw[(0 * 16 + r + 8 * hi) * 32 + lo]      = acc00[r] * inv0;
    sOw[(1 * 16 + r + 8 * hi) * 32 + lo]      = acc01[r] * inv0;
    sOw[(2 * 16 + r + 8 * hi) * 32 + lo]      = acc02[r] * inv0;
    sOw[(3 * 16 + r + 8 * hi) * 32 + lo]      = acc03[r] * inv0;
    sOw[(0 * 16 + r + 8 * hi) * 32 + 16 + lo] = acc10[r] * inv1;
    sOw[(1 * 16 + r + 8 * hi) * 32 + 16 + lo] = acc11[r] * inv1;
    sOw[(2 * 16 + r + 8 * hi) * 32 + 16 + lo] = acc12[r] * inv1;
    sOw[(3 * 16 + r + 8 * hi) * 32 + 16 + lo] = acc13[r] * inv1;
  }

  // B-fragments of O: two cv chunks x two n-tiles
  v16bf bo0a, bo0b, bo1a, bo1b;
#pragma unroll
  for (int t = 0; t < 16; ++t) {
    int kk = t + 16 * hi;
    bo0a[t] = (__bf16)sOw[kk * 32 + lo];
    bo0b[t] = (__bf16)sOw[(32 + kk) * 32 + lo];
    bo1a[t] = (__bf16)sOw[kk * 32 + 16 + lo];
    bo1b[t] = (__bf16)sOw[(32 + kk) * 32 + 16 + lo];
  }

  const float g = gamma[0];
#pragma unroll
  for (int ct = 0; ct < 8; ++ct) {
    v16bf aw0, aw1;
#pragma unroll
    for (int t = 0; t < 16; ++t) {
      int kk = t + 16 * hi;
      aw0[t] = sWo[(ct * 16 + lo) * CV + kk];
      aw1[t] = sWo[(ct * 16 + lo) * CV + 32 + kk];
    }
    v8f y0 = WMMA_BF16(aw0, bo0a, zero);
    y0 = WMMA_BF16(aw1, bo0b, y0);
    v8f y1 = WMMA_BF16(aw0, bo1a, zero);
    y1 = WMMA_BF16(aw1, bo1b, y1);
#pragma unroll
    for (int r = 0; r < 8; ++r) {
      int row = ct * 16 + r + 8 * hi;
      size_t oi = (size_t)(b * CC + row) * NQ + n0 + lo;
      out[oi]      = g * y0[r] + x[oi];
      out[oi + 16] = g * y1[r] + x[oi + 16];
    }
  }
}

// ---------------------------------------------------------------------------
// Launch. Workspace layout (all offsets 256B-aligned, total ~23.5 MB):
//   [0)         wbf     96*128  bf16 (24576 B)
//   [24576)     wobf    128*64  bf16 (16384 B)
//   [40960)     projall 8*96*13824 bf16 (21233664 B)
//   [21274624)  kbuf    8*16*1728 bf16 (442368 B)
//   [21716992)  vbuf    8*64*1728 bf16 (1769472 B)
// ---------------------------------------------------------------------------
extern "C" void kernel_launch(void* const* d_in, const int* in_sizes, int n_in,
                              void* d_out, int out_size, void* d_ws, size_t ws_size,
                              hipStream_t stream) {
  const float* x  = (const float*)d_in[0];
  const float* wq = (const float*)d_in[1];
  const float* wk = (const float*)d_in[2];
  const float* wv = (const float*)d_in[3];
  const float* wo = (const float*)d_in[4];
  const float* gm = (const float*)d_in[5];
  float* out = (float*)d_out;

  char* ws = (char*)d_ws;
  __bf16* wbf     = (__bf16*)(ws + 0);
  __bf16* wobf    = (__bf16*)(ws + 24576);
  __bf16* projall = (__bf16*)(ws + 40960);
  __bf16* kbuf    = (__bf16*)(ws + 21274624);
  __bf16* vbuf    = (__bf16*)(ws + 21716992);

  attn3d_wconv<<<40, 512, 0, stream>>>(wq, wk, wv, wo, wbf, wobf);

  attn3d_proj<<<dim3(NQ / 128, BB), 128, 0, stream>>>(x, wbf, projall);

  {
    int total = BB * 80 * NK;
    attn3d_pool<<<(total + 255) / 256, 256, 0, stream>>>(projall, kbuf, vbuf);
  }

  attn3d_flash<<<dim3(NQ / 128, BB), 128, 0, stream>>>(projall, kbuf, vbuf, wobf, x, gm, out);
}
